// Model_31413390803603
// MI455X (gfx1250) — matrix operand order repaired and checked by reading, not compiled, not run on hardware
//
#include <hip/hip_runtime.h>
#include <hip/hip_bf16.h>
#include <cstdint>
#include <cstddef>

// ---------------------------------------------------------------------------
// Model dims (from reference)
// ---------------------------------------------------------------------------
#define B_      2
#define S_      1024
#define E_      512
#define INNER_  1024
#define NH_     8
#define DH_     128
#define NB_     4
#define NHP_    256
#define KCONV_  4
#define PRED_   96
#define COUT_   21
#define TOK_    (B_*S_)          // 2048 tokens
#define EPS_    1e-5f

typedef __attribute__((ext_vector_type(16))) _Float16 v16h;
typedef __attribute__((ext_vector_type(8)))  _Float16 v8h;
typedef __attribute__((ext_vector_type(8)))  float    v8f;

union AFrag { v16h v; v8h h[2]; };

// ---------------------------------------------------------------------------
// Weight convert + transpose: f32 [K,N] row-major -> f16 [N,K]
// ---------------------------------------------------------------------------
__global__ void k_cvt_t(const float* __restrict__ src, _Float16* __restrict__ dst,
                        int K, int N) {
  int idx = blockIdx.x * 256 + threadIdx.x;
  if (idx >= K * N) return;
  int k = idx / N, n = idx % N;
  dst[(size_t)n * K + k] = (_Float16)src[idx];
}

// ---------------------------------------------------------------------------
// Embedding: x[t,e] = concat(x_enc, x_mark)[t,:25] @ emb_W + emb_b
// ---------------------------------------------------------------------------
__global__ void k_embed(const float* __restrict__ xe, const float* __restrict__ xm,
                        const float* __restrict__ W, const float* __restrict__ bia,
                        float* __restrict__ x) {
  int idx = blockIdx.x * 256 + threadIdx.x;
  if (idx >= TOK_ * E_) return;
  int t = idx / E_, e = idx % E_;
  float acc = bia[e];
  const float* pe = xe + (size_t)t * 21;
  const float* pm = xm + (size_t)t * 4;
#pragma unroll
  for (int i = 0; i < 21; i++) acc += pe[i] * W[i * E_ + e];
#pragma unroll
  for (int i = 0; i < 4;  i++) acc += pm[i] * W[(21 + i) * E_ + e];
  x[idx] = acc;
}

// ---------------------------------------------------------------------------
// LayerNorm over E=512, scale by w, emit f16 (GEMM A operand)
// ---------------------------------------------------------------------------
__global__ void k_ln16(const float* __restrict__ x, const float* __restrict__ w,
                       _Float16* __restrict__ out) {
  int t = blockIdx.x, tid = threadIdx.x;       // 256 threads
  const float* xp = x + (size_t)t * E_;
  float v0 = xp[tid], v1 = xp[tid + 256];
  float s = v0 + v1, ss = v0 * v0 + v1 * v1;
#pragma unroll
  for (int m = 16; m >= 1; m >>= 1) { s += __shfl_xor(s, m, 32); ss += __shfl_xor(ss, m, 32); }
  __shared__ float sh[16];
  int wid = tid >> 5, lane = tid & 31;
  if (lane == 0) { sh[wid] = s; sh[8 + wid] = ss; }
  __syncthreads();
  float st = 0.f, sst = 0.f;
#pragma unroll
  for (int i = 0; i < 8; i++) { st += sh[i]; sst += sh[8 + i]; }
  float mu = st / E_;
  float var = sst / E_ - mu * mu;
  float r = rsqrtf(var + EPS_);
  out[(size_t)t * E_ + tid]       = (_Float16)((v0 - mu) * r * w[tid]);
  out[(size_t)t * E_ + tid + 256] = (_Float16)((v1 - mu) * r * w[tid + 256]);
}

// ---------------------------------------------------------------------------
// WMMA GEMM: C[M,N] = A[M,K](f16) @ Wt[N,K](f16)^T  (+ optional f32 residual)
// One wave computes a 32x64 macro-tile: 2x4 grid of 16x16 WMMA tiles.
// Per K=32 step: 2 A-frags + 4 B-frags feed 8 v_wmma_f32_16x16x32_f16
// (1.5 b128 loads per WMMA).  ~150 live VGPRs -> no scratch spills; paired
// with __launch_bounds__(32,1) so the allocator gets the full VGPR budget.
// Fragment layouts per ISA 7.12.2 (wave32):
//   A: lane m=L%16, halves j<8 -> K=8g+j, j>=8 -> K=16+8g+(j-8)
//   B: lane n=L%16, halves j<8 -> K=8g+j, j>=8 -> K=16+8g+(j-8)
// ---------------------------------------------------------------------------
__global__ void __launch_bounds__(32, 1)
k_gemm(const _Float16* __restrict__ A, const _Float16* __restrict__ Bt,
       const float* __restrict__ resid, float* __restrict__ C,
       int M, int N, int K) {
  int tile = blockIdx.x;
  int nt = N >> 6;                      // 64-wide macro tiles
  int tm = tile / nt, tn = tile % nt;
  int lane = threadIdx.x & 31;
  int g = lane >> 4, m16 = lane & 15;

  const _Float16* ap = A  + (size_t)(tm * 32 + m16) * K;   // +mi*16*K per row tile
  const _Float16* bp = Bt + (size_t)(tn * 64 + m16) * K;   // +ni*16*K per col tile

  v8f acc[2][4];
#pragma unroll
  for (int mi = 0; mi < 2; mi++)
#pragma unroll
    for (int ni = 0; ni < 4; ni++)
      acc[mi][ni] = (v8f){0.f,0.f,0.f,0.f,0.f,0.f,0.f,0.f};

  for (int k0 = 0; k0 < K; k0 += 32) {
    __builtin_prefetch(ap + k0 + 64, 0, 1);     // global_prefetch_b8
    __builtin_prefetch(bp + k0 + 64, 0, 1);
    AFrag a[2], b[4];
#pragma unroll
    for (int mi = 0; mi < 2; mi++) {
      const _Float16* p = ap + (size_t)mi * 16 * K + k0;
      a[mi].h[0] = *(const v8h*)(p + 8 * g);
      a[mi].h[1] = *(const v8h*)(p + 16 + 8 * g);
    }
#pragma unroll
    for (int ni = 0; ni < 4; ni++) {
      const _Float16* p = bp + (size_t)ni * 16 * K + k0;
      b[ni].h[0] = *(const v8h*)(p + 8 * g);
      b[ni].h[1] = *(const v8h*)(p + 16 + 8 * g);
    }
#pragma unroll
    for (int mi = 0; mi < 2; mi++)
#pragma unroll
      for (int ni = 0; ni < 4; ni++)
        acc[mi][ni] = __builtin_amdgcn_wmma_f32_16x16x32_f16(
            false, a[mi].v, false, b[ni].v, (short)0, acc[mi][ni], false, false);
  }

#pragma unroll
  for (int mi = 0; mi < 2; mi++)
#pragma unroll
    for (int ni = 0; ni < 4; ni++)
#pragma unroll
      for (int r = 0; r < 8; r++) {
        int row = tm * 32 + mi * 16 + r + 8 * g;
        size_t o = (size_t)row * N + tn * 64 + ni * 16 + m16;
        float val = acc[mi][ni][r];
        if (resid) val += resid[o];
        C[o] = val;
      }
}

// ---------------------------------------------------------------------------
// Causal depthwise conv (K=4) on xm = up[..., :INNER], + bias, SiLU
// ---------------------------------------------------------------------------
__global__ void k_conv(const float* __restrict__ up, const float* __restrict__ cw,
                       const float* __restrict__ cb, float* __restrict__ xc) {
  int idx = blockIdx.x * 256 + threadIdx.x;
  if (idx >= TOK_ * INNER_) return;
  int c = idx % INNER_;
  int t = idx / INNER_;
  int b = t / S_, s = t % S_;
  float acc = cb[c];
#pragma unroll
  for (int k = 0; k < KCONV_; k++) {
    int ss = s - (KCONV_ - 1) + k;
    if (ss >= 0)
      acc += up[((size_t)(b * S_ + ss)) * (2 * INNER_) + c] * cw[c * KCONV_ + k];
  }
  xc[idx] = acc / (1.f + __expf(-acc));   // SiLU
}

// ---------------------------------------------------------------------------
// Headwise 4x4 mixes: q,k from xc; v from xm. Emit f16 in [B*NH, S, DH]
// (row-major) plus V transposed [B*NH, DH, S] for the P@V WMMA B-operand.
// ---------------------------------------------------------------------------
__global__ void k_qkv(const float* __restrict__ xc, const float* __restrict__ up,
                      const float* __restrict__ qW, const float* __restrict__ kW,
                      const float* __restrict__ vW,
                      _Float16* __restrict__ Q, _Float16* __restrict__ Km,
                      _Float16* __restrict__ V, _Float16* __restrict__ Vt) {
  int idx = blockIdx.x * 256 + threadIdx.x;
  if (idx >= TOK_ * INNER_) return;
  int c = idx % INNER_, t = idx / INNER_;
  int b = t / S_, s = t % S_;
  int hg = c >> 2, o = c & 3;
  const float* xcp = xc + (size_t)t * INNER_ + (hg << 2);
  const float* xmp = up + (size_t)t * (2 * INNER_) + (hg << 2);
  float q = 0.f, kk = 0.f, vv = 0.f;
#pragma unroll
  for (int d = 0; d < 4; d++) {
    int wi = (hg * 4 + o) * 4 + d;
    q  += xcp[d] * qW[wi];
    kk += xcp[d] * kW[wi];
    vv += xmp[d] * vW[wi];
  }
  int n = c >> 7, dh = c & 127;
  size_t qi = (((size_t)(b * NH_ + n)) * S_ + s) * DH_ + dh;
  Q[qi]  = (_Float16)q;
  Km[qi] = (_Float16)kk;
  V[qi]  = (_Float16)vv;
  Vt[((size_t)(b * NH_ + n) * DH_ + dh) * S_ + s] = (_Float16)vv;
}

// ---------------------------------------------------------------------------
// Gate projections: ig/fg[b,h,s] = qkv[t,:3072] @ W[:,h] + b[h]
// ---------------------------------------------------------------------------
__global__ void k_gates(const _Float16* __restrict__ Q, const _Float16* __restrict__ Km,
                        const _Float16* __restrict__ V,
                        const float* __restrict__ igW, const float* __restrict__ igb,
                        const float* __restrict__ fgW, const float* __restrict__ fgb,
                        float* __restrict__ igo, float* __restrict__ fgo) {
  int t = blockIdx.x;
  int b = t / S_, s = t % S_;
  int tid = threadIdx.x;
  int head = tid >> 5, lane = tid & 31;
  float si = 0.f, sf = 0.f;
  for (int c = lane; c < 3 * INNER_; c += 32) {
    int p = c >> 10, cc = c & 1023;
    int n = cc >> 7, d = cc & 127;
    size_t qi = (((size_t)(b * NH_ + n)) * S_ + s) * DH_ + d;
    float v = (p == 0) ? (float)Q[qi] : (p == 1) ? (float)Km[qi] : (float)V[qi];
    si += v * igW[c * NH_ + head];
    sf += v * fgW[c * NH_ + head];
  }
#pragma unroll
  for (int m = 16; m >= 1; m >>= 1) { si += __shfl_xor(si, m, 32); sf += __shfl_xor(sf, m, 32); }
  if (lane == 0) {
    igo[((size_t)(b * NH_ + head)) * S_ + s] = si + igb[head];
    fgo[((size_t)(b * NH_ + head)) * S_ + s] = sf + fgb[head];
  }
}

// ---------------------------------------------------------------------------
// Sequential scan: cs = cumsum(log_sigmoid(fg)) along S, per (b,h)
// ---------------------------------------------------------------------------
__global__ void k_cumlogsig(const float* __restrict__ fg, float* __restrict__ cs) {
  int t = blockIdx.x * blockDim.x + threadIdx.x;
  if (t >= B_ * NH_) return;
  float run = 0.f;
  for (int s = 0; s < S_; s++) {
    float f = fg[(size_t)t * S_ + s];
    float ls = (f > 0.f) ? -log1pf(__expf(-f)) : (f - log1pf(__expf(f)));
    run += ls;
    cs[(size_t)t * S_ + s] = run;
  }
}

// ---------------------------------------------------------------------------
// mLSTM parallel attention, flash-style per 16-row query tile, one wave each.
// Processes 32 keys per iteration: two score WMMA chains (sc0: keys kt..kt+15,
// sc1: keys kt+16..kt+31) then one full-K=32 P@V pass (all halves real).
// logD = cs[q]-cs[t]+ig[t] (causal); running rowmax M, signed rowsum L;
// h = (sum_t qk*exp(logD-M) * v) / (max(|L|, exp(-M)) + 1e-6)
// ---------------------------------------------------------------------------
__global__ void __launch_bounds__(32, 1)
k_attn(const _Float16* __restrict__ Q, const _Float16* __restrict__ Km,
       const _Float16* __restrict__ Vt, const float* __restrict__ cs,
       const float* __restrict__ ig, float* __restrict__ Hout) {
  const float QKS = 0.08838834764831845f;   // 1/sqrt(DH)
  int wid = blockIdx.x;
  int qt = wid % (S_ / 16);
  int bh = wid / (S_ / 16);
  int b = bh / NH_, n = bh % NH_;
  int q0 = qt * 16;
  int lane = threadIdx.x & 31;
  int g = lane >> 4, m16 = lane & 15;

  __shared__ __align__(16) _Float16 lds[16 * 32];   // P staging, 16 rows x 32 keys

  // preload q A-fragments for the 4 dh chunks
  AFrag qa[4];
  const _Float16* qp = Q + ((size_t)bh * S_ + q0 + m16) * DH_;
#pragma unroll
  for (int d = 0; d < 4; d++) {
    qa[d].h[0] = *(const v8h*)(qp + d * 32 + 8 * g);
    qa[d].h[1] = *(const v8h*)(qp + d * 32 + 16 + 8 * g);
  }
  float csq[8];
#pragma unroll
  for (int r = 0; r < 8; r++) csq[r] = cs[(size_t)bh * S_ + q0 + r + 8 * g];

  float M8[8], L8[8];
  v8f accN[8];
#pragma unroll
  for (int r = 0; r < 8; r++) { M8[r] = -1e30f; L8[r] = 0.f; }
#pragma unroll
  for (int d = 0; d < 8; d++) accN[d] = (v8f){0.f,0.f,0.f,0.f,0.f,0.f,0.f,0.f};

  for (int kt = 0; kt <= q0; kt += 32) {
    // --- raw scores over two 16-key sub-tiles ---
    int krow1 = kt + 16 + m16;
    int krow1c = (krow1 < S_) ? krow1 : (S_ - 1);   // clamp (masked anyway)
    const _Float16* kp0 = Km + ((size_t)bh * S_ + kt + m16) * DH_;
    const _Float16* kp1 = Km + ((size_t)bh * S_ + krow1c) * DH_;
    v8f sc0 = {0.f,0.f,0.f,0.f,0.f,0.f,0.f,0.f};
    v8f sc1 = {0.f,0.f,0.f,0.f,0.f,0.f,0.f,0.f};
#pragma unroll
    for (int d = 0; d < 4; d++) {
      AFrag kb;
      kb.h[0] = *(const v8h*)(kp0 + d * 32 + 8 * g);
      kb.h[1] = *(const v8h*)(kp0 + d * 32 + 16 + 8 * g);
      sc0 = __builtin_amdgcn_wmma_f32_16x16x32_f16(false, qa[d].v, false, kb.v,
                                                   (short)0, sc0, false, false);
    }
#pragma unroll
    for (int d = 0; d < 4; d++) {
      AFrag kb;
      kb.h[0] = *(const v8h*)(kp1 + d * 32 + 8 * g);
      kb.h[1] = *(const v8h*)(kp1 + d * 32 + 16 + 8 * g);
      sc1 = __builtin_amdgcn_wmma_f32_16x16x32_f16(false, qa[d].v, false, kb.v,
                                                   (short)0, sc1, false, false);
    }
    float csk0 = cs[(size_t)bh * S_ + kt + m16];
    float igk0 = ig[(size_t)bh * S_ + kt + m16];
    float csk1 = cs[(size_t)bh * S_ + krow1c];
    float igk1 = ig[(size_t)bh * S_ + krow1c];
    float scl[8];
#pragma unroll
    for (int r = 0; r < 8; r++) {
      int qrow = q0 + r + 8 * g;
      bool mk0 = (kt + m16) > qrow;
      bool mk1 = (kt + 16 + m16) > qrow;
      float lr0 = mk0 ? -1e30f : (csq[r] - csk0 + igk0);
      float lr1 = mk1 ? -1e30f : (csq[r] - csk1 + igk1);
      float tmax = fmaxf(lr0, lr1);
      tmax = fmaxf(tmax, __shfl_xor(tmax, 1, 32));
      tmax = fmaxf(tmax, __shfl_xor(tmax, 2, 32));
      tmax = fmaxf(tmax, __shfl_xor(tmax, 4, 32));
      tmax = fmaxf(tmax, __shfl_xor(tmax, 8, 32));
      float nm = fmaxf(M8[r], tmax);
      scl[r] = __expf(M8[r] - nm);
      M8[r] = nm;
      float p0 = mk0 ? 0.f : sc0[r] * QKS * __expf(lr0 - nm);
      float p1 = mk1 ? 0.f : sc1[r] * QKS * __expf(lr1 - nm);
      float ps = p0 + p1;
      ps += __shfl_xor(ps, 1, 32); ps += __shfl_xor(ps, 2, 32);
      ps += __shfl_xor(ps, 4, 32); ps += __shfl_xor(ps, 8, 32);
      L8[r] = L8[r] * scl[r] + ps;
      lds[(r + 8 * g) * 32 + m16]      = (_Float16)p0;
      lds[(r + 8 * g) * 32 + 16 + m16] = (_Float16)p1;
    }
    __builtin_amdgcn_wave_barrier();
    // P as A fragment (16x32, all 32 keys real)
    AFrag pa;
    pa.h[0] = *(const v8h*)(&lds[m16 * 32 + 16 * g]);
    pa.h[1] = *(const v8h*)(&lds[m16 * 32 + 16 * g + 8]);
    int kb2 = kt + 16 * g;
    if (kb2 > S_ - 16) kb2 = S_ - 16;   // never triggers for S=1024, kept for safety
#pragma unroll
    for (int d = 0; d < 8; d++) {
      AFrag vb;
      const _Float16* vp = Vt + ((size_t)bh * DH_ + d * 16 + m16) * S_;
      vb.h[0] = *(const v8h*)(vp + kb2);
      vb.h[1] = *(const v8h*)(vp + kb2 + 8);
      v8f tacc = accN[d];
#pragma unroll
      for (int r = 0; r < 8; r++) tacc[r] *= scl[r];
      accN[d] = __builtin_amdgcn_wmma_f32_16x16x32_f16(false, pa.v, false, vb.v,
                                                       (short)0, tacc, false, false);
    }
  }
  // finalize: norm = max(|L|, exp(-M)) + 1e-6
  float inv[8];
#pragma unroll
  for (int r = 0; r < 8; r++) {
    float norm = fmaxf(fabsf(L8[r]), __expf(-M8[r]));
    inv[r] = 1.f / (norm + 1e-6f);
  }
#pragma unroll
  for (int d = 0; d < 8; d++) {
#pragma unroll
    for (int r = 0; r < 8; r++) {
      int q = q0 + r + 8 * g;
      int dh = d * 16 + m16;
      Hout[((size_t)(b * S_ + q)) * INNER_ + n * DH_ + dh] = accN[d][r] * inv[r];
    }
  }
}

// ---------------------------------------------------------------------------
// Per-head LN over DH, * onorm_w, + skip*xc, * silu(z); emit f16 hs.
// ---------------------------------------------------------------------------
__global__ void k_mix(const float* __restrict__ h, const float* __restrict__ xc,
                      const float* __restrict__ up, const float* __restrict__ ow,
                      const float* __restrict__ sk, _Float16* __restrict__ hs) {
  int t = blockIdx.x;
  int tid = threadIdx.x;
  int head = tid >> 5, lane = tid & 31;
  const float* hp = h + (size_t)t * INNER_ + head * DH_;
  float v[4]; float s = 0.f, ss = 0.f;
#pragma unroll
  for (int i = 0; i < 4; i++) { v[i] = hp[lane * 4 + i]; s += v[i]; ss += v[i] * v[i]; }
#pragma unroll
  for (int m = 16; m >= 1; m >>= 1) { s += __shfl_xor(s, m, 32); ss += __shfl_xor(ss, m, 32); }
  float mu = s / DH_;
  float var = ss / DH_ - mu * mu;
  float r = rsqrtf(var + EPS_);
#pragma unroll
  for (int i = 0; i < 4; i++) {
    int c = head * DH_ + lane * 4 + i;
    float z = up[(size_t)t * (2 * INNER_) + INNER_ + c];
    float sz = z / (1.f + __expf(-z));
    float hn = (v[i] - mu) * r * ow[c];
    hs[(size_t)t * INNER_ + c] = (_Float16)((hn + sk[c] * xc[(size_t)t * INNER_ + c]) * sz);
  }
}

// ---------------------------------------------------------------------------
// Final: post-LN then head projection on last PRED_ steps
// ---------------------------------------------------------------------------
__global__ void k_head(const float* __restrict__ x, const float* __restrict__ w,
                       const float* __restrict__ hW, const float* __restrict__ hb,
                       float* __restrict__ out) {
  int blk = blockIdx.x;
  int b = blk / PRED_, p = blk % PRED_;
  int s = S_ - PRED_ + p;
  int tid = threadIdx.x;
  const float* xp = x + ((size_t)(b * S_ + s)) * E_;
  float v0 = xp[tid], v1 = xp[tid + 256];
  float sum = v0 + v1, sq = v0 * v0 + v1 * v1;
#pragma unroll
  for (int m = 16; m >= 1; m >>= 1) { sum += __shfl_xor(sum, m, 32); sq += __shfl_xor(sq, m, 32); }
  __shared__ float sh[16];
  __shared__ float xn[E_];
  int wid = tid >> 5, lane = tid & 31;
  if (lane == 0) { sh[wid] = sum; sh[8 + wid] = sq; }
  __syncthreads();
  float st = 0.f, sst = 0.f;
#pragma unroll
  for (int i = 0; i < 8; i++) { st += sh[i]; sst += sh[8 + i]; }
  float mu = st / E_, var = sst / E_ - mu * mu, r = rsqrtf(var + EPS_);
  xn[tid] = (v0 - mu) * r * w[tid];
  xn[tid + 256] = (v1 - mu) * r * w[tid + 256];
  __syncthreads();
  if (tid < COUT_) {
    float acc = hb[tid];
    for (int e = 0; e < E_; e++) acc += xn[e] * hW[e * COUT_ + tid];
    out[((size_t)(b * PRED_ + p)) * COUT_ + tid] = acc;
  }
}

// ---------------------------------------------------------------------------
// Orchestration
// ---------------------------------------------------------------------------
extern "C" void kernel_launch(void* const* d_in, const int* in_sizes, int n_in,
                              void* d_out, int out_size, void* d_ws, size_t ws_size,
                              hipStream_t stream) {
  (void)in_sizes; (void)n_in; (void)out_size; (void)ws_size;

  const float* x_enc   = (const float*)d_in[0];
  const float* x_mark  = (const float*)d_in[1];
  // d_in[2], d_in[3] (x_dec, x_mark_dec) unused by reference math
  const float* emb_W   = (const float*)d_in[4];
  const float* emb_b   = (const float*)d_in[5];
  const float* ln_w    = (const float*)d_in[6];
  const float* up_W    = (const float*)d_in[7];
  const float* conv_W  = (const float*)d_in[8];
  const float* conv_b  = (const float*)d_in[9];
  const float* q_W     = (const float*)d_in[10];
  const float* k_W     = (const float*)d_in[11];
  const float* v_W     = (const float*)d_in[12];
  const float* ig_W    = (const float*)d_in[13];
  const float* ig_b    = (const float*)d_in[14];
  const float* fg_W    = (const float*)d_in[15];
  const float* fg_b    = (const float*)d_in[16];
  const float* skip    = (const float*)d_in[17];
  const float* onorm_w = (const float*)d_in[18];
  const float* down_W  = (const float*)d_in[19];
  const float* post_ln = (const float*)d_in[20];
  const float* head_W  = (const float*)d_in[21];
  const float* head_b  = (const float*)d_in[22];
  float* out = (float*)d_out;

  // ---- workspace carve (256B aligned) ----
  char* wsp = (char*)d_ws;
  auto carve = [&](size_t bytes) -> void* {
    void* p = (void*)wsp;
    wsp += (bytes + 255) & ~(size_t)255;
    return p;
  };
  _Float16* upWt = (_Float16*)carve((size_t)NB_ * 2 * INNER_ * E_ * 2);   // [NB][2048][512]
  _Float16* dnWt = (_Float16*)carve((size_t)NB_ * E_ * INNER_ * 2);       // [NB][512][1024]
  float*    xbuf = (float*)   carve((size_t)TOK_ * E_ * 4);
  _Float16* xn16 = (_Float16*)carve((size_t)TOK_ * E_ * 2);
  float*    upb  = (float*)   carve((size_t)TOK_ * 2 * INNER_ * 4);
  float*    xcb  = (float*)   carve((size_t)TOK_ * INNER_ * 4);
  _Float16* Qh   = (_Float16*)carve((size_t)B_ * NH_ * S_ * DH_ * 2);
  _Float16* Kh   = (_Float16*)carve((size_t)B_ * NH_ * S_ * DH_ * 2);
  _Float16* Vh   = (_Float16*)carve((size_t)B_ * NH_ * S_ * DH_ * 2);
  _Float16* Vth  = (_Float16*)carve((size_t)B_ * NH_ * DH_ * S_ * 2);
  float*    igbf = (float*)   carve((size_t)B_ * NH_ * S_ * 4);
  float*    fgbf = (float*)   carve((size_t)B_ * NH_ * S_ * 4);
  float*    csbf = (float*)   carve((size_t)B_ * NH_ * S_ * 4);
  float*    hbuf = (float*)   carve((size_t)TOK_ * INNER_ * 4);
  _Float16* hs16 = (_Float16*)carve((size_t)TOK_ * INNER_ * 2);

  // ---- weight conversion (f32 [K,N] -> f16 [N,K]) ----
  for (int i = 0; i < NB_; i++) {
    int kn1 = E_ * 2 * INNER_;
    k_cvt_t<<<(kn1 + 255) / 256, 256, 0, stream>>>(
        up_W + (size_t)i * kn1, upWt + (size_t)i * kn1, E_, 2 * INNER_);
    int kn2 = INNER_ * E_;
    k_cvt_t<<<(kn2 + 255) / 256, 256, 0, stream>>>(
        down_W + (size_t)i * kn2, dnWt + (size_t)i * kn2, INNER_, E_);
  }

  // ---- embedding ----
  k_embed<<<(TOK_ * E_ + 255) / 256, 256, 0, stream>>>(x_enc, x_mark, emb_W, emb_b, xbuf);

  // ---- blocks ----
  for (int i = 0; i < NB_; i++) {
    k_ln16<<<TOK_, 256, 0, stream>>>(xbuf, ln_w + (size_t)i * E_, xn16);

    // up GEMM: [2048,512] x [512,2048], 32x64 macro-tiles
    k_gemm<<<(TOK_ / 32) * (2 * INNER_ / 64), 32, 0, stream>>>(
        xn16, upWt + (size_t)i * (2 * INNER_) * E_, nullptr, upb,
        TOK_, 2 * INNER_, E_);

    k_conv<<<(TOK_ * INNER_ + 255) / 256, 256, 0, stream>>>(
        upb, conv_W + (size_t)i * INNER_ * KCONV_, conv_b + (size_t)i * INNER_, xcb);

    k_qkv<<<(TOK_ * INNER_ + 255) / 256, 256, 0, stream>>>(
        xcb, upb,
        q_W + (size_t)i * NHP_ * 16, k_W + (size_t)i * NHP_ * 16, v_W + (size_t)i * NHP_ * 16,
        Qh, Kh, Vh, Vth);

    k_gates<<<TOK_, 256, 0, stream>>>(
        Qh, Kh, Vh,
        ig_W + (size_t)i * 3 * INNER_ * NH_, ig_b + (size_t)i * NH_,
        fg_W + (size_t)i * 3 * INNER_ * NH_, fg_b + (size_t)i * NH_,
        igbf, fgbf);

    k_cumlogsig<<<1, 32, 0, stream>>>(fgbf, csbf);

    k_attn<<<(S_ / 16) * (B_ * NH_), 32, 0, stream>>>(Qh, Kh, Vth, csbf, igbf, hbuf);

    k_mix<<<TOK_, 256, 0, stream>>>(
        hbuf, xcb, upb, onorm_w + (size_t)i * INNER_, skip + (size_t)i * INNER_, hs16);

    // down GEMM with residual, in-place on xbuf (each element touched once)
    k_gemm<<<(TOK_ / 32) * (E_ / 64), 32, 0, stream>>>(
        hs16, dnWt + (size_t)i * E_ * INNER_, xbuf, xbuf, TOK_, E_, INNER_);
  }

  // ---- post-LN + head ----
  k_head<<<B_ * PRED_, 256, 0, stream>>>(xbuf, post_ln, head_W, head_b, out);
}
